// self_Qwen3VLTextAttention_67293547594096
// MI455X (gfx1250) — compile-verified, matrix operand order checked by reading, not run on hardware
//
#include <hip/hip_runtime.h>
#include <cstdint>
#include <cstddef>

typedef __attribute__((ext_vector_type(16))) __bf16 v16bf;
typedef __attribute__((ext_vector_type(8)))  float  v8f;

#define S_LEN 4096
#define HID_D 2048
#define HQ_N  16
#define HKV_N 4
#define DH_D  128
#define QKV_N 3072
#define SCALE_F 0.08838834764831845f   // 1/sqrt(128)

// ---------------------------------------------------------------------------
// f32 -> bf16 conversion
// ---------------------------------------------------------------------------
__global__ void cvt_bf16_kernel(const float* __restrict__ x, __bf16* __restrict__ y, int n)
{
    int i = blockIdx.x * 256 + threadIdx.x;
    if (i < n) y[i] = (__bf16)x[i];
}

// W[K][N] (row-major) -> Wt[N][K] bf16 (so B fragments are contiguous loads)
__global__ void transpose_cvt_kernel(const float* __restrict__ W, __bf16* __restrict__ Wt,
                                     int K, int N)
{
    int idx = blockIdx.x * 256 + threadIdx.x;
    if (idx >= K * N) return;
    int k = idx / N, n = idx - k * N;
    Wt[(size_t)n * K + k] = (__bf16)W[idx];
}

// ---------------------------------------------------------------------------
// GEMM: C[M,N] f32 = A[M,K] bf16 * Bt[N,K] bf16   (Bt[n][k] == B[k][n])
// block = 256 threads (8 waves, 2x4 wave grid), block tile 128x256,
// wave tile 64x64: 16 WMMAs per 16 b128 loads per 32-deep k-step.
// ---------------------------------------------------------------------------
__global__ __launch_bounds__(256) void gemm_bf16_wmma(const __bf16* __restrict__ A,
                                                      const __bf16* __restrict__ Bt,
                                                      float* __restrict__ C,
                                                      int M, int N, int K)
{
    const int w    = threadIdx.x >> 5;
    const int lane = threadIdx.x & 31;
    const int wm = w & 1;          // 2 wave rows
    const int wn = w >> 1;         // 4 wave cols
    const int lm = lane & 15;
    const int half = lane >> 4;
    const int kh = half << 4;      // K offset within 32-chunk per lane half

    const int rowBase = blockIdx.y * 128 + wm * 64;
    const int colBase = blockIdx.x * 256 + wn * 64;

    const __bf16* ap[4];
#pragma unroll
    for (int r = 0; r < 4; ++r)
        ap[r] = A + (size_t)(rowBase + r * 16 + lm) * K + kh;
    const __bf16* bp[4];
#pragma unroll
    for (int c = 0; c < 4; ++c)
        bp[c] = Bt + (size_t)(colBase + c * 16 + lm) * K + kh;

    v8f acc[4][4] = {};

    for (int kk = 0; kk < K; kk += 32) {
        __builtin_prefetch(ap[0] + kk + 256, 0, 1);
        __builtin_prefetch(bp[0] + kk + 256, 0, 1);
        v16bf a[4], b[4];
#pragma unroll
        for (int r = 0; r < 4; ++r) a[r] = *(const v16bf*)(ap[r] + kk);
#pragma unroll
        for (int c = 0; c < 4; ++c) b[c] = *(const v16bf*)(bp[c] + kk);
#pragma unroll
        for (int r = 0; r < 4; ++r)
#pragma unroll
            for (int c = 0; c < 4; ++c)
                acc[r][c] = __builtin_amdgcn_wmma_f32_16x16x32_bf16(
                    false, a[r], false, b[c], (short)0, acc[r][c], false, false);
    }

#pragma unroll
    for (int r = 0; r < 4; ++r)
#pragma unroll
        for (int c = 0; c < 4; ++c)
#pragma unroll
            for (int e = 0; e < 8; ++e)
                C[(size_t)(rowBase + r * 16 + e + 8 * half) * N
                  + (colBase + c * 16 + lm)] = acc[r][c][e];
}

// ---------------------------------------------------------------------------
// RMSNorm + RoPE (Q, K) and V repack. grid = (S, 24): y<16 Q-head, 16..19 K,
// 20..23 V. block = 128 threads = one 128-wide head row.
// Q out: [h][s][d], K out: [h][s][d], V out transposed: [h][d][s]
// ---------------------------------------------------------------------------
__global__ __launch_bounds__(128) void normrope_kernel(const float* __restrict__ qkv,
    const float* __restrict__ cosb, const float* __restrict__ sinb,
    const float* __restrict__ qnw,  const float* __restrict__ knw,
    __bf16* __restrict__ Qb, __bf16* __restrict__ Kb, __bf16* __restrict__ Vt)
{
    const int s = blockIdx.x;
    const int y = blockIdx.y;
    const int d = threadIdx.x;
    __shared__ float vsh[DH_D];
    __shared__ float part[4];

    if (y >= 20) {  // V: plain convert + transpose
        int hk = y - 20;
        float val = qkv[(size_t)s * QKV_N + 2560 + hk * DH_D + d];
        Vt[((size_t)hk * DH_D + d) * S_LEN + s] = (__bf16)val;
        return;
    }

    int col; const float* wn; __bf16* outp;
    if (y < 16) { col = y * DH_D;               wn = qnw; outp = Qb + ((size_t)y * S_LEN + s) * DH_D; }
    else        { int hk = y - 16;
                  col = 2048 + hk * DH_D;       wn = knw; outp = Kb + ((size_t)hk * S_LEN + s) * DH_D; }

    float x = qkv[(size_t)s * QKV_N + col + d];
    float ss = x * x;
#pragma unroll
    for (int m = 16; m >= 1; m >>= 1) ss += __shfl_xor(ss, m, 32);
    if ((threadIdx.x & 31) == 0) part[threadIdx.x >> 5] = ss;
    __syncthreads();
    float tot = part[0] + part[1] + part[2] + part[3];
    float rms = rsqrtf(tot * (1.0f / DH_D) + 1e-6f);
    float nv  = x * rms * wn[d];
    vsh[d] = nv;
    __syncthreads();
    float rot = (d < 64) ? -vsh[d + 64] : vsh[d - 64];
    float o = nv * cosb[(size_t)s * DH_D + d] + rot * sinb[(size_t)s * DH_D + d];
    outp[d] = (__bf16)o;
}

// ---------------------------------------------------------------------------
// Causal flash attention. grid = (S/64, HQ), block = 128 (4 waves).
// Each wave: 16 query rows, iterates 64-key blocks with online softmax.
// ---------------------------------------------------------------------------
__global__ __launch_bounds__(128) void flash_attn_kernel(const __bf16* __restrict__ Qb,
    const __bf16* __restrict__ Kb, const __bf16* __restrict__ Vt,
    __bf16* __restrict__ attnB)
{
    const int h    = blockIdx.y;
    const int w    = threadIdx.x >> 5;
    const int lane = threadIdx.x & 31;
    const int lm   = lane & 15;
    const int half = lane >> 4;
    const int kh   = half << 4;
    const int qr0  = blockIdx.x * 64 + w * 16;
    const int hk   = h >> 2;  // GQA: 4 q-heads per kv head

    const __bf16* Qh = Qb + ((size_t)h * S_LEN + qr0) * DH_D;
    const __bf16* Kh = Kb + (size_t)hk * S_LEN * DH_D;
    const __bf16* Vh = Vt + (size_t)hk * DH_D * S_LEN;   // [d][s]

    __shared__ __align__(32) __bf16 Plds[4][16][64];     // per-wave P staging

    v16bf qa[4];
#pragma unroll
    for (int ks = 0; ks < 4; ++ks)
        qa[ks] = *(const v16bf*)(Qh + (size_t)lm * DH_D + ks * 32 + kh);

    v8f o[8] = {};
    float mrow[8], lrow[8];
#pragma unroll
    for (int e = 0; e < 8; ++e) { mrow[e] = -1e30f; lrow[e] = 0.f; }

    for (int kb = 0; kb <= qr0 + 15; kb += 64) {
        // ---- S = Q K^T for 16 q-rows x 64 keys ----
        v8f sacc[4] = {};
#pragma unroll
        for (int ks = 0; ks < 4; ++ks) {
#pragma unroll
            for (int c = 0; c < 4; ++c) {
                v16bf kf = *(const v16bf*)(Kh + (size_t)(kb + c * 16 + lm) * DH_D + ks * 32 + kh);
                sacc[c] = __builtin_amdgcn_wmma_f32_16x16x32_bf16(
                    false, qa[ks], false, kf, (short)0, sacc[c], false, false);
            }
        }
        // ---- mask + online softmax (rows e+8*half; 16-lane row reductions) ----
#pragma unroll
        for (int e = 0; e < 8; ++e) {
            const int m    = e + 8 * half;
            const int qrow = qr0 + m;
            float sv[4];
            float lmax = -1e30f;
#pragma unroll
            for (int c = 0; c < 4; ++c) {
                int key = kb + c * 16 + lm;
                float s = sacc[c][e] * SCALE_F;
                s = (key <= qrow) ? s : -1e30f;
                sv[c] = s;
                lmax = fmaxf(lmax, s);
            }
#pragma unroll
            for (int mk = 1; mk < 16; mk <<= 1) lmax = fmaxf(lmax, __shfl_xor(lmax, mk, 32));
            float mnew  = fmaxf(mrow[e], lmax);
            float alpha = __expf(mrow[e] - mnew);
            float psum  = 0.f;
#pragma unroll
            for (int c = 0; c < 4; ++c) {
                float p = __expf(sv[c] - mnew);
                psum += p;
                Plds[w][m][c * 16 + lm] = (__bf16)p;
            }
#pragma unroll
            for (int mk = 1; mk < 16; mk <<= 1) psum += __shfl_xor(psum, mk, 32);
            lrow[e] = lrow[e] * alpha + psum;
            mrow[e] = mnew;
#pragma unroll
            for (int dt = 0; dt < 8; ++dt) o[dt][e] *= alpha;
        }
        // lanes exchange P through LDS within one wave: wait on DScnt, no barrier
        asm volatile("s_wait_dscnt 0" ::: "memory");
        // ---- O += P V ----
#pragma unroll
        for (int kk2 = 0; kk2 < 2; ++kk2) {
            v16bf pa = *(const v16bf*)&Plds[w][lm][kk2 * 32 + kh];
#pragma unroll
            for (int dt = 0; dt < 8; ++dt) {
                v16bf vb = *(const v16bf*)(Vh + (size_t)(dt * 16 + lm) * S_LEN + kb + kk2 * 32 + kh);
                o[dt] = __builtin_amdgcn_wmma_f32_16x16x32_bf16(
                    false, pa, false, vb, (short)0, o[dt], false, false);
            }
        }
    }

    // ---- normalize + write attn in [s][h*128+d] layout (bf16 for O-proj GEMM) ----
#pragma unroll
    for (int e = 0; e < 8; ++e) {
        const int m = e + 8 * half;
        float inv = 1.0f / lrow[e];
#pragma unroll
        for (int dt = 0; dt < 8; ++dt) {
            float val = o[dt][e] * inv;
            attnB[(size_t)(qr0 + m) * HID_D + h * DH_D + dt * 16 + lm] = (__bf16)val;
        }
    }
}

// ---------------------------------------------------------------------------
// Host launch
// ---------------------------------------------------------------------------
extern "C" void kernel_launch(void* const* d_in, const int* in_sizes, int n_in,
                              void* d_out, int out_size, void* d_ws, size_t ws_size,
                              hipStream_t stream)
{
    (void)in_sizes; (void)n_in; (void)out_size; (void)ws_size;
    const float* hidden = (const float*)d_in[0];
    const float* cosb   = (const float*)d_in[1];
    const float* sinb   = (const float*)d_in[2];
    const float* wq     = (const float*)d_in[3];
    const float* wk     = (const float*)d_in[4];
    const float* wv     = (const float*)d_in[5];
    const float* wo     = (const float*)d_in[6];
    const float* qnw    = (const float*)d_in[7];
    const float* knw    = (const float*)d_in[8];

    size_t off = 0;
    auto carve = [&](size_t bytes) -> void* {
        void* p = (char*)d_ws + off;
        off += (bytes + 255) & ~(size_t)255;
        return p;
    };
    __bf16* hb    = (__bf16*)carve((size_t)S_LEN * HID_D * 2);          // hidden bf16
    __bf16* wqkvt = (__bf16*)carve((size_t)QKV_N * HID_D * 2);          // [3072][2048]
    __bf16* wot   = (__bf16*)carve((size_t)HID_D * HID_D * 2);          // [2048][2048]
    float*  qkvf  = (float*) carve((size_t)S_LEN * QKV_N * 4);          // QKV f32
    __bf16* Qb    = (__bf16*)carve((size_t)HQ_N  * S_LEN * DH_D * 2);   // [h][s][d]
    __bf16* Kb    = (__bf16*)carve((size_t)HKV_N * S_LEN * DH_D * 2);   // [h][s][d]
    __bf16* Vt    = (__bf16*)carve((size_t)HKV_N * DH_D * S_LEN * 2);   // [h][d][s]
    __bf16* attnB = (__bf16*)carve((size_t)S_LEN * HID_D * 2);          // [s][2048]

    // 1) precision conversion / weight repack
    cvt_bf16_kernel<<<(S_LEN * HID_D + 255) / 256, 256, 0, stream>>>(hidden, hb, S_LEN * HID_D);
    transpose_cvt_kernel<<<(HID_D * 2048 + 255) / 256, 256, 0, stream>>>(wq, wqkvt, HID_D, 2048);
    transpose_cvt_kernel<<<(HID_D * 512  + 255) / 256, 256, 0, stream>>>(wk, wqkvt + (size_t)2048 * HID_D, HID_D, 512);
    transpose_cvt_kernel<<<(HID_D * 512  + 255) / 256, 256, 0, stream>>>(wv, wqkvt + (size_t)2560 * HID_D, HID_D, 512);
    transpose_cvt_kernel<<<(HID_D * 2048 + 255) / 256, 256, 0, stream>>>(wo, wot, HID_D, 2048);

    // 2) fused QKV projection (WMMA bf16), block tile 128x256
    gemm_bf16_wmma<<<dim3(QKV_N / 256, S_LEN / 128), 256, 0, stream>>>(
        hb, wqkvt, qkvf, S_LEN, QKV_N, HID_D);

    // 3) RMSNorm + RoPE + V repack
    normrope_kernel<<<dim3(S_LEN, 24), 128, 0, stream>>>(
        qkvf, cosb, sinb, qnw, knw, Qb, Kb, Vt);

    // 4) causal flash attention (WMMA bf16)
    flash_attn_kernel<<<dim3(S_LEN / 64, HQ_N), 128, 0, stream>>>(Qb, Kb, Vt, attnB);

    // 5) output projection (WMMA bf16 -> f32 out)
    gemm_bf16_wmma<<<dim3(HID_D / 256, S_LEN / 128), 256, 0, stream>>>(
        attnB, wot, (float*)d_out, S_LEN, HID_D, HID_D);
}
